// MPNN_49280454754409
// MI455X (gfx1250) — matrix operand order repaired and checked by reading, not compiled, not run on hardware
//
#include <hip/hip_runtime.h>
#include <cmath>

typedef __bf16 bf16;
typedef __attribute__((ext_vector_type(16))) __bf16 v16bf;
typedef __attribute__((ext_vector_type(8)))  float  v8f;
typedef int v4i __attribute__((vector_size(16)));

#define WMMA_BF16 __builtin_amdgcn_wmma_f32_16x16x32_bf16

#define HDIM 32
#define NODE_IN 64
#define EDGE_IN 16
#define EDGE_H 128
#define NCHUNK 129                     // 128 main K-chunks (4 k0 x 32 h) + 1 bias chunk
#define B_LDS_BYTES (NCHUNK * 2 * 32 * 16 * 2)   // 264192 bytes of packed B fragments
#define OUT_STRIDE 40                            // halves per staged row (80 B, 16B-aligned)
#define OUTS_BYTES (8 * 16 * OUT_STRIDE * 2)     // 10240
#define DSTS_BYTES (8 * 16 * 4)
#define MSG_LDS (B_LDS_BYTES + OUTS_BYTES + DSTS_BYTES)

// ---------------------------------------------------------------- utilities
__global__ void k_zero(float* __restrict__ p, long n) {
    long i = (long)blockIdx.x * blockDim.x + threadIdx.x;
    long stride = (long)gridDim.x * blockDim.x;
    for (; i < n; i += stride) p[i] = 0.0f;
}

// ------------------------------------------------- out0 = relu(nf @ W0^T + b0)
__global__ void k_lin0(const float* __restrict__ nf, const float* __restrict__ w,
                       const float* __restrict__ bias, float* __restrict__ out,
                       bf16* __restrict__ outb, int N) {
    __shared__ float ws[HDIM * NODE_IN];
    __shared__ float bs[HDIM];
    for (int i = threadIdx.x; i < HDIM * NODE_IN; i += blockDim.x) ws[i] = w[i];
    if (threadIdx.x < HDIM) bs[threadIdx.x] = bias[threadIdx.x];
    __syncthreads();
    int idx = blockIdx.x * blockDim.x + threadIdx.x;
    if (idx >= N * HDIM) return;
    int n = idx >> 5, o = idx & 31;
    float s = bs[o];
    const float* row = nf + (long)n * NODE_IN;
    const float* wr  = ws + o * NODE_IN;
#pragma unroll 8
    for (int i = 0; i < NODE_IN; ++i) s += row[i] * wr[i];
    s = fmaxf(s, 0.0f);
    out[idx] = s;
    outb[idx] = (bf16)s;
}

// ------------------------------------------- r1 = bf16(relu(ef @ W1^T + b1))
__global__ void k_edge1(const float* __restrict__ ef, const float* __restrict__ w,
                        const float* __restrict__ bias, bf16* __restrict__ r1, int E) {
    __shared__ float ws[EDGE_H * EDGE_IN];
    __shared__ float bs[EDGE_H];
    for (int i = threadIdx.x; i < EDGE_H * EDGE_IN; i += blockDim.x) ws[i] = w[i];
    for (int i = threadIdx.x; i < EDGE_H; i += blockDim.x) bs[i] = bias[i];
    __syncthreads();
    int idx = blockIdx.x * blockDim.x + threadIdx.x;
    if (idx >= E * EDGE_H) return;
    int e = idx >> 7, k = idx & 127;
    float s = bs[k];
    const float* row = ef + (long)e * EDGE_IN;
    const float* wr  = ws + k * EDGE_IN;
#pragma unroll
    for (int i = 0; i < EDGE_IN; ++i) s += row[i] * wr[i];
    r1[idx] = (bf16)fmaxf(s, 0.0f);
}

// ------- Pack V[(h,k),o] = en2_w[h*32+o, k] (+ bias chunk) into WMMA B-fragment
// order: element ((chunk*2 + t)*32 + lane)*16 + j, with o = t*16 + (lane&15),
// K = j<8 ? (lane>>4)*8 + j : 16 + (lane>>4)*8 + (j-8) (B layout matches A's K runs).
__global__ void k_pack_V(const float* __restrict__ en2_w, const float* __restrict__ en2_b,
                         bf16* __restrict__ Vp) {
    int idx = blockIdx.x * blockDim.x + threadIdx.x;
    if (idx >= NCHUNK * 1024) return;
    int j     = idx & 15;
    int l     = (idx >> 4) & 31;
    int t     = (idx >> 9) & 1;
    int chunk = idx >> 10;
    int o = t * 16 + (l & 15);
    int K = (j < 8) ? (l >> 4) * 8 + j : 16 + (l >> 4) * 8 + (j - 8);
    float v;
    if (chunk < 128) {
        int k0i = chunk >> 5;     // chunk = k0i*32 + h
        int h   = chunk & 31;
        int k   = k0i * 32 + K;
        v = en2_w[(h * 32 + o) * 128 + k];
    } else {
        v = en2_b[K * 32 + o];    // B2[h',o] = en2_b[h'*32+o], K == h'
    }
    Vp[idx] = (bf16)v;
}

// --------------------------------------------- fused edge GEMM + scatter-add
// msg = U @ V where U[e,(h,k)] = out[src[e],h] * r1[e,k] built on the fly.
// One wave = M-tile of 16 edges; V resident in LDS (258 KB); persistent WGs.
__global__ void __launch_bounds__(256)
k_msg(const bf16* __restrict__ outb, const bf16* __restrict__ r1,
      const bf16* __restrict__ Vp, const int* __restrict__ src,
      const int* __restrict__ dst, float* __restrict__ agg,
      int E, int nGroups) {
    extern __shared__ char smem[];
    bf16* Bs   = (bf16*)smem;
    bf16* outS = (bf16*)(smem + B_LDS_BYTES);
    int*  dstS = (int*)(smem + B_LDS_BYTES + OUTS_BYTES);

    const int tid  = threadIdx.x;
    const int wave = tid >> 5;
    const int lane = tid & 31;

    // Stage entire packed V (+bias) matrix into LDS once; prefer the gfx1250
    // async global->LDS path when the toolchain exposes it.
#if __has_builtin(__builtin_amdgcn_global_load_async_to_lds_b128) && \
    __has_builtin(__builtin_amdgcn_s_wait_asynccnt)
    {
        v4i* g = (v4i*)Vp;            // global source, 16B granules
        v4i* l = (v4i*)Bs;            // LDS destination, 16B granules
        for (int i = tid; i < B_LDS_BYTES / 16; i += blockDim.x)
            __builtin_amdgcn_global_load_async_to_lds_b128(g + i, l + i, 0, 0);
        __builtin_amdgcn_s_wait_asynccnt(0);
    }
#else
    {
        const uint4* g = (const uint4*)Vp;
        uint4* s = (uint4*)Bs;
        for (int i = tid; i < B_LDS_BYTES / 16; i += blockDim.x) s[i] = g[i];
    }
#endif
    __syncthreads();

    bf16* myOutS = outS + wave * (16 * OUT_STRIDE);
    int*  myDstS = dstS + wave * 16;
    const int mrow = lane & 15;
    const int krun = lane >> 4;              // 0 or 1: which K half-run this lane owns
    const int nTiles = (E + 15) >> 4;

    union U16 { uint4 q[2]; v16bf v; };

    for (int grp = blockIdx.x; grp < nGroups; grp += gridDim.x) {
        const int tile = grp * 8 + wave;     // wave-uniform; EXEC stays full
        if (tile >= nTiles) continue;
        const int ebase = tile * 16;

        // Stage the 16 gathered out[src] rows (bf16, padded stride) and dst ids;
        // wave-local -> only a dscnt wait is needed, no block barrier.
        for (int j = 0; j < 16; ++j) {
            int e = ebase + j; if (e >= E) e = E - 1;
            int row = src[e];
            myOutS[j * OUT_STRIDE + lane] = outb[row * HDIM + lane];
            if (lane == 0) myDstS[j] = dst[e];
        }
        asm volatile("s_wait_dscnt 0x0" ::: "memory");

        v8f acc0 = {0.f, 0.f, 0.f, 0.f, 0.f, 0.f, 0.f, 0.f};
        v8f acc1 = {0.f, 0.f, 0.f, 0.f, 0.f, 0.f, 0.f, 0.f};

        int em = ebase + mrow; if (em >= E) em = E - 1;
        const bf16* r1row = r1 + (long)em * EDGE_H;

        for (int k0i = 0; k0i < 4; ++k0i) {
            const int k0 = k0i * 32;
            // A-layout K runs per lane: {krun*8..+7} and {16+krun*8..+7}
            U16 r1u;
            r1u.q[0] = *(const uint4*)(r1row + k0 + krun * 8);
            r1u.q[1] = *(const uint4*)(r1row + k0 + 16 + krun * 8);
            const v16bf r1v = r1u.v;
#pragma unroll
            for (int h = 0; h < 32; ++h) {
                const int chunk = k0i * 32 + h;
                const bf16 ab = myOutS[mrow * OUT_STRIDE + h];
                v16bf af;
#pragma unroll
                for (int j = 0; j < 16; ++j) af[j] = (bf16)(r1v[j] * ab);
                const v16bf b0 = *(const v16bf*)(Bs + ((chunk * 2 + 0) * 32 + lane) * 16);
                const v16bf b1 = *(const v16bf*)(Bs + ((chunk * 2 + 1) * 32 + lane) * 16);
                acc0 = WMMA_BF16(false, af, false, b0, (short)0, acc0, false, false);
                acc1 = WMMA_BF16(false, af, false, b1, (short)0, acc1, false, false);
            }
        }
        {   // bias chunk: msg += out_src @ B2  (A rows = staged out, two aligned b128s)
            U16 au;
            const bf16* pb = myOutS + mrow * OUT_STRIDE + krun * 8;
            au.q[0] = *(const uint4*)pb;
            au.q[1] = *(const uint4*)(pb + 16);
            const v16bf af = au.v;
            const v16bf b0 = *(const v16bf*)(Bs + ((128 * 2 + 0) * 32 + lane) * 16);
            const v16bf b1 = *(const v16bf*)(Bs + ((128 * 2 + 1) * 32 + lane) * 16);
            acc0 = WMMA_BF16(false, af, false, b0, (short)0, acc0, false, false);
            acc1 = WMMA_BF16(false, af, false, b1, (short)0, acc1, false, false);
        }
        // C/D layout: reg r, lanes 0-15 -> (M=r, N=lane); lanes 16-31 -> (M=r+8)
#pragma unroll
        for (int r = 0; r < 8; ++r) {
            const int m = r + 8 * krun;
            if (ebase + m < E) {
                const int d = myDstS[m];
                unsafeAtomicAdd(&agg[d * HDIM + (lane & 15)],      acc0[r]);
                unsafeAtomicAdd(&agg[d * HDIM + 16 + (lane & 15)], acc1[r]);
            }
        }
    }
}

// -------------------- per-node: conv/residual/lin1/relu update (one thread/(n,o))
__global__ void k_update(const float* __restrict__ agg, const float* __restrict__ cur,
                         const float* __restrict__ h0, const float* __restrict__ lin1_w,
                         const float* __restrict__ lin1_b, const float* __restrict__ conv_bias,
                         float* __restrict__ nxt, bf16* __restrict__ outb, int N) {
    __shared__ float w[HDIM * HDIM];
    __shared__ float b[HDIM], cb[HDIM];
    for (int i = threadIdx.x; i < HDIM * HDIM; i += blockDim.x) w[i] = lin1_w[i];
    if (threadIdx.x < HDIM) { b[threadIdx.x] = lin1_b[threadIdx.x]; cb[threadIdx.x] = conv_bias[threadIdx.x]; }
    __syncthreads();
    int idx = blockIdx.x * blockDim.x + threadIdx.x;
    if (idx >= N * HDIM) return;
    int n = idx >> 5, o = idx & 31;
    float s = 0.0f, to = 0.0f;
#pragma unroll 8
    for (int i = 0; i < HDIM; ++i) {
        float conv = agg[n * HDIM + i] + cur[n * HDIM + i] + cb[i];
        float t = 0.5f * conv + 0.5f * h0[n * HDIM + i];
        s += t * w[o * HDIM + i];
        if (i == o) to = t;
    }
    float v = (1.0f / 3.0f) * (s + b[o]) + (2.0f / 3.0f) * to;
    v = fmaxf(v, 0.0f);
    nxt[idx] = v;
    outb[idx] = (bf16)v;
}

// ------------------------------------------------ batchnorm statistics (sum, sumsq)
__global__ void k_bn_stats(const float* __restrict__ x, float* __restrict__ stats, int N) {
    __shared__ float ls[64];
    if (threadIdx.x < 64) ls[threadIdx.x] = 0.0f;
    __syncthreads();
    const int c = threadIdx.x & 31;
    float s = 0.0f, q = 0.0f;
    for (long i = (long)blockIdx.x * blockDim.x + threadIdx.x; i < (long)N * HDIM;
         i += (long)gridDim.x * blockDim.x) {
        float v = x[i];     // channel == c (stride is a multiple of 32)
        s += v; q += v * v;
    }
    atomicAdd(&ls[c], s);
    atomicAdd(&ls[32 + c], q);
    __syncthreads();
    if (threadIdx.x < 64) unsafeAtomicAdd(&stats[threadIdx.x], ls[threadIdx.x]);
}

__global__ void k_bn_finalize(float* __restrict__ stats, const float* __restrict__ gamma,
                              const float* __restrict__ beta, int N) {
    int c = threadIdx.x;
    if (c >= HDIM) return;
    float mu  = stats[c] / (float)N;
    float var = stats[32 + c] / (float)N - mu * mu;
    float scale = gamma[c] * rsqrtf(var + 1e-5f);
    stats[c] = scale;
    stats[32 + c] = beta[c] - mu * scale;
}

// ----------------------------- node head: y_bn + sigmoid(y_bn @ ylin_w^T + b)
__global__ void k_head_node(const float* __restrict__ x, const float* __restrict__ stats,
                            const float* __restrict__ yw, const float* __restrict__ yb,
                            float* __restrict__ out, float* __restrict__ ybn, int N) {
    int n = blockIdx.x * blockDim.x + threadIdx.x;
    if (n >= N) return;
    float v[HDIM];
#pragma unroll 8
    for (int i = 0; i < HDIM; ++i) {
        float y = x[n * HDIM + i] * stats[i] + stats[32 + i];
        v[i] = y;
        ybn[n * HDIM + i] = y;
    }
#pragma unroll
    for (int c = 0; c < 3; ++c) {
        float s = yb[c];
#pragma unroll 8
        for (int i = 0; i < HDIM; ++i) s += v[i] * yw[c * HDIM + i];
        out[n * 3 + c] = 1.0f / (1.0f + expf(-s));
    }
}

// --------------------- edge head: sigmoid((ybn[s]*ybn[d]) @ ylin2_w^T + b)
__global__ void k_head_edge(const float* __restrict__ ybn, const int* __restrict__ sl,
                            const int* __restrict__ dl, const float* __restrict__ w2,
                            const float* __restrict__ b2, float* __restrict__ out, int E2) {
    int e = blockIdx.x * blockDim.x + threadIdx.x;
    if (e >= E2) return;
    const float* a = ybn + (long)sl[e] * HDIM;
    const float* b = ybn + (long)dl[e] * HDIM;
    float s = b2[0];
#pragma unroll 8
    for (int i = 0; i < HDIM; ++i) s += a[i] * b[i] * w2[i];
    out[e] = 1.0f / (1.0f + expf(-s));
}

// ============================================================== launch glue
extern "C" void kernel_launch(void* const* d_in, const int* in_sizes, int n_in,
                              void* d_out, int out_size, void* d_ws, size_t ws_size,
                              hipStream_t stream) {
    const float* n_feat    = (const float*)d_in[0];
    const float* e_feat    = (const float*)d_in[1];
    const int*   src       = (const int*)d_in[2];
    const int*   dst       = (const int*)d_in[3];
    const int*   src_list  = (const int*)d_in[4];
    const int*   dst_list  = (const int*)d_in[5];
    const float* lin0_w    = (const float*)d_in[6];
    const float* lin0_b    = (const float*)d_in[7];
    const float* en1_w     = (const float*)d_in[8];
    const float* en1_b     = (const float*)d_in[9];
    const float* en2_w     = (const float*)d_in[10];
    const float* en2_b     = (const float*)d_in[11];
    const float* conv_bias = (const float*)d_in[12];
    const float* lin1_w    = (const float*)d_in[13];
    const float* lin1_b    = (const float*)d_in[14];
    const float* bn_gamma  = (const float*)d_in[15];
    const float* bn_beta   = (const float*)d_in[16];
    const float* ylin_w    = (const float*)d_in[17];
    const float* ylin_b    = (const float*)d_in[18];
    const float* ylin2_w   = (const float*)d_in[19];
    const float* ylin2_b   = (const float*)d_in[20];

    const int N  = in_sizes[0] / NODE_IN;
    const int E  = in_sizes[2];
    const int E2 = in_sizes[4];

    char* ws = (char*)d_ws;
    size_t off = 0;
    auto alloc = [&](size_t bytes) -> void* {
        void* p = ws + off;
        off = (off + bytes + 255) & ~(size_t)255;
        return p;
    };
    float* outA  = (float*)alloc((size_t)N * HDIM * 4);   // also serves as h0 (never overwritten)
    float* outB  = (float*)alloc((size_t)N * HDIM * 4);
    float* outC  = (float*)alloc((size_t)N * HDIM * 4);
    bf16*  outb16= (bf16*) alloc((size_t)N * HDIM * 2);
    float* agg   = (float*)alloc((size_t)N * HDIM * 4);
    bf16*  r1    = (bf16*) alloc((size_t)E * EDGE_H * 2);
    bf16*  Vp    = (bf16*) alloc((size_t)NCHUNK * 1024 * 2);
    float* stats = (float*)alloc(64 * 4);
    float* ybn   = (float*)alloc((size_t)N * HDIM * 4);

    const int B = 256;
    k_lin0 <<<(N * HDIM + B - 1) / B, B, 0, stream>>>(n_feat, lin0_w, lin0_b, outA, outb16, N);
    k_edge1<<<(E * EDGE_H + B - 1) / B, B, 0, stream>>>(e_feat, en1_w, en1_b, r1, E);
    k_pack_V<<<(NCHUNK * 1024 + B - 1) / B, B, 0, stream>>>(en2_w, en2_b, Vp);

    const int nTiles  = (E + 15) / 16;
    const int nGroups = (nTiles + 7) / 8;
    const float* cur = outA;
    float* nxt = outB;
    for (int step = 0; step < 3; ++step) {
        k_zero<<<256, B, 0, stream>>>(agg, (long)N * HDIM);
        k_msg<<<256, B, MSG_LDS, stream>>>(outb16, r1, Vp, src, dst, agg, E, nGroups);
        k_update<<<(N * HDIM + B - 1) / B, B, 0, stream>>>(agg, cur, outA, lin1_w, lin1_b,
                                                           conv_bias, nxt, outb16, N);
        cur = nxt;
        nxt = (nxt == outB) ? outC : outB;
    }

    k_zero<<<1, 64, 0, stream>>>(stats, 64);
    k_bn_stats<<<256, B, 0, stream>>>(cur, stats, N);
    k_bn_finalize<<<1, 32, 0, stream>>>(stats, bn_gamma, bn_beta, N);

    float* out_nodes = (float*)d_out;             // (N,3) then (E2,1), flat
    float* out_edges = out_nodes + (size_t)N * 3;
    k_head_node<<<(N + B - 1) / B, B, 0, stream>>>(cur, stats, ylin_w, ylin_b, out_nodes, ybn, N);
    k_head_edge<<<(E2 + B - 1) / B, B, 0, stream>>>(ybn, src_list, dst_list, ylin2_w, ylin2_b,
                                                    out_edges, E2);
}
